// Model_12438225289616
// MI455X (gfx1250) — matrix operand order repaired and checked by reading, not compiled, not run on hardware
//
#include <hip/hip_runtime.h>
#include <hip/hip_bf16.h>
#include <math.h>

typedef __attribute__((ext_vector_type(16))) __bf16 v16bf;
typedef __attribute__((ext_vector_type(16))) short  v16s;
typedef __attribute__((ext_vector_type(8)))  short  short8;
typedef __attribute__((ext_vector_type(8)))  float  v8f;

#define EMB   300
#define KPAD  320
#define KT    (KPAD / 32)   // 10 K-steps
#define HID   256
#define GATE  1024          // 4*HID
#define QT    64
#define RROWS 64            // 8 + 24 + 8 + 24

// ---------- helpers ----------
__device__ __forceinline__ unsigned short f2bf(float f) {
    unsigned int u = __float_as_uint(f);
    unsigned int r = u + 0x7FFFu + ((u >> 16) & 1u);   // round-to-nearest-even
    return (unsigned short)(r >> 16);
}

__device__ __forceinline__ float sigmoidf_dev(float x) {
    return 1.0f / (1.0f + __expf(-x));
}

// ---------- phase 1a: embedding gather + f32->bf16, pad K 300->320 ----------
__global__ __launch_bounds__(KPAD)
void gather_kernel(const int* __restrict__ qtok,
                   const int* __restrict__ prel, const int* __restrict__ pwrel,
                   const int* __restrict__ nrel, const int* __restrict__ nwrel,
                   const float* __restrict__ qemb, const float* __restrict__ remb,
                   unsigned short* __restrict__ xq, unsigned short* __restrict__ xr) {
    int b = blockIdx.x, tid = threadIdx.x;
    int token; const float* emb; unsigned short* dst;
    if (b < QT) {
        token = qtok[b]; emb = qemb; dst = xq + (size_t)b * KPAD;
    } else {
        int rb = b - QT; emb = remb; dst = xr + (size_t)rb * KPAD;
        if      (rb < 8)  token = prel[rb];
        else if (rb < 32) token = pwrel[rb - 8];
        else if (rb < 40) token = nrel[rb - 32];
        else              token = nwrel[rb - 40];
    }
    float v = (tid < EMB) ? emb[(size_t)token * EMB + tid] : 0.0f;
    dst[tid] = f2bf(v);
}

// ---------- phase 1b: W_ih [1024,300] -> bf16 B-fragments [g][nt][kt][lane][16] ----------
// Pre-swizzle into the WMMA B-operand lane layout this code uses (contiguous k order):
//   lane holds N = nt*16 + (lane&15); element e -> K = kt*32 + e + (lane>=16 ? 16 : 0).
// GEMM then loads one contiguous 32B chunk per lane (wave: contiguous 1KB fragment).
__global__ __launch_bounds__(KT * 32)
void wswz_kernel(const float* __restrict__ w_qf, const float* __restrict__ w_qb,
                 const float* __restrict__ w_rf, const float* __restrict__ w_rb,
                 unsigned short* __restrict__ wt) {
    int nt = blockIdx.x;              // 0..63 (N tile)
    int g  = blockIdx.y;              // 0..3  (qf, qb, rf, rb)
    int kt   = threadIdx.x >> 5;      // 0..9
    int lane = threadIdx.x & 31;
    int half = lane >> 4, l15 = lane & 15;

    const float* W = (g == 0) ? w_qf : (g == 1) ? w_qb : (g == 2) ? w_rf : w_rb;
    const float* src = W + (size_t)(nt * 16 + l15) * EMB;   // row j of W_ih
    unsigned short* dst = wt + ((((size_t)g * 64 + nt) * KT + kt) * 32 + lane) * 16;

    int kbase = kt * 32 + half * 16;
#pragma unroll
    for (int e = 0; e < 16; ++e) {
        int k = kbase + e;
        dst[e] = f2bf((k < EMB) ? src[k] : 0.0f);
    }
}

// ---------- phase 1c: bias pre-sums  b_ih + b_hh  per (lstm,dir) ----------
__global__ __launch_bounds__(256)
void bias_kernel(const float* a0, const float* b0, const float* a1, const float* b1,
                 const float* a2, const float* b2, const float* a3, const float* b3,
                 float* __restrict__ bsum) {
    int g = blockIdx.x;
    const float* A = (g == 0) ? a0 : (g == 1) ? a1 : (g == 2) ? a2 : a3;
    const float* B = (g == 0) ? b0 : (g == 1) ? b1 : (g == 2) ? b2 : b3;
    for (int j = threadIdx.x; j < GATE; j += 256)
        bsum[g * GATE + j] = A[j] + B[j];
}

// ---------- phase 2: xg = x @ W_ih^T + bias  (4 GEMMs, M=64 K=320 N=1024, bf16 WMMA) ----------
// One wave per 16x16 output tile; 10 fully-unrolled K-steps of v_wmma_f32_16x16x32_bf16.
// A fragment (16-bit A 16x32, contiguous k order as in B): lane holds M=lane&15;
//   elems 0..15 -> K = kbase + half*16 + 0..15 (contiguous 32B).
// B fragment: pre-swizzled contiguous 32B per lane (see wswz_kernel).
__global__ __launch_bounds__(32)
void xg_gemm_kernel(const __bf16* __restrict__ xq, const __bf16* __restrict__ xr,
                    const __bf16* __restrict__ wt,   // fragment-major B
                    const float*  __restrict__ bsum, // [4][1024]
                    float* __restrict__ xg) {        // [4][64][1024]
    int nt   = blockIdx.x;     // N tile 0..63
    int mt   = blockIdx.y;     // M tile 0..3
    int g    = blockIdx.z;     // GEMM id 0..3
    int lane = threadIdx.x;
    int half = lane >> 4;
    int l15  = lane & 15;

    const __bf16* X = (g < 2) ? xq : xr;
    const __bf16* arow  = X + (size_t)(mt * 16 + l15) * KPAD + half * 16;
    const __bf16* bbase = wt + (((size_t)g * 64 + nt) * KT * 32 + lane) * 16;

    v8f c = {0.f, 0.f, 0.f, 0.f, 0.f, 0.f, 0.f, 0.f};
#pragma unroll
    for (int kt = 0; kt < KT; ++kt) {
        const short8* ap = (const short8*)(arow + kt * 32);      // 16B aligned
        short8 alo = ap[0];          // K offsets +0..+7
        short8 ahi = ap[1];          // K offsets +8..+15
        const short8* bp = (const short8*)(bbase + (size_t)kt * 32 * 16);
        short8 blo = bp[0];
        short8 bhi = bp[1];
        v16s as = __builtin_shufflevector(alo, ahi, 0, 1, 2, 3, 4, 5, 6, 7,
                                          8, 9, 10, 11, 12, 13, 14, 15);
        v16s bs = __builtin_shufflevector(blo, bhi, 0, 1, 2, 3, 4, 5, 6, 7,
                                          8, 9, 10, 11, 12, 13, 14, 15);
        c = __builtin_amdgcn_wmma_f32_16x16x32_bf16(false, __builtin_bit_cast(v16bf, as),
                                                    false, __builtin_bit_cast(v16bf, bs),
                                                    (short)0, c, false, false);
    }

    const float* bias = bsum + g * GATE;
    float* out = xg + (size_t)g * 64 * GATE;
    int col = nt * 16 + l15;
    float bcol = bias[col];
#pragma unroll
    for (int r = 0; r < 8; ++r) {
        int row = mt * 16 + r + half * 8;   // C/D layout: vgpr r, lanes 16..31 -> M=r+8
        out[(size_t)row * GATE + col] = c[r] + bcol;
    }
}

// ---------- phase 3: 10 independent LSTM scans (latency-bound, VALU matvec) ----------
// Thread n owns hidden unit n and gate rows {n, n+256, n+512, n+768} of W_hh.
// h lives in LDS (b128 broadcast reads); W_hh rows stream from L2 as b128.
__global__ __launch_bounds__(256)
void lstm_scan_kernel(const float* __restrict__ xg,      // [4][64][1024]
                      const float* __restrict__ whh_qf, const float* __restrict__ whh_qb,
                      const float* __restrict__ whh_rf, const float* __restrict__ whh_rb,
                      float* __restrict__ q_out,         // [64][512]
                      float* __restrict__ r_out) {       // [64][512]
    static const int g_tab[10]    = {0, 1, 2, 3, 2, 3, 2, 3, 2, 3};
    static const int row0_tab[10] = {0, 0, 0, 0, 8, 8, 32, 32, 40, 40};
    static const int T_tab[10]    = {64, 64, 8, 8, 24, 24, 8, 8, 24, 24};
    static const int rev_tab[10]  = {0, 1, 0, 1, 0, 1, 0, 1, 0, 1};

    int s    = blockIdx.x;
    int g    = g_tab[s];
    int row0 = row0_tab[s];
    int T    = T_tab[s];
    int rev  = rev_tab[s];
    int tid  = threadIdx.x;

    const float* Whh = (g == 0) ? whh_qf : (g == 1) ? whh_qb : (g == 2) ? whh_rf : whh_rb;
    float* outb = (s < 2) ? q_out : r_out;
    int colOff  = rev * HID;       // bwd direction -> cols 256..511

    const float4* W0 = (const float4*)(Whh + (size_t)(0 * HID + tid) * HID);
    const float4* W1 = (const float4*)(Whh + (size_t)(1 * HID + tid) * HID);
    const float4* W2 = (const float4*)(Whh + (size_t)(2 * HID + tid) * HID);
    const float4* W3 = (const float4*)(Whh + (size_t)(3 * HID + tid) * HID);

    __shared__ float h[HID];
    h[tid] = 0.0f;
    float c = 0.0f;
    __syncthreads();
    const float4* h4 = (const float4*)h;

    for (int step = 0; step < T; ++step) {
        int t = rev ? (T - 1 - step) : step;
        const float* xr = xg + ((size_t)g * 64 + row0 + t) * GATE;
        // prefetch next step's gate row while we crunch this one
        if (step + 1 < T) {
            int tn = rev ? (T - 2 - step) : (step + 1);
            __builtin_prefetch(xg + ((size_t)g * 64 + row0 + tn) * GATE + tid, 0, 0);
        }

        float g0 = xr[0 * HID + tid];
        float g1 = xr[1 * HID + tid];
        float g2 = xr[2 * HID + tid];
        float g3 = xr[3 * HID + tid];
#pragma unroll 4
        for (int k4 = 0; k4 < HID / 4; ++k4) {
            float4 hv = h4[k4];
            float4 w0 = W0[k4], w1 = W1[k4], w2 = W2[k4], w3 = W3[k4];
            g0 = fmaf(w0.x, hv.x, fmaf(w0.y, hv.y, fmaf(w0.z, hv.z, fmaf(w0.w, hv.w, g0))));
            g1 = fmaf(w1.x, hv.x, fmaf(w1.y, hv.y, fmaf(w1.z, hv.z, fmaf(w1.w, hv.w, g1))));
            g2 = fmaf(w2.x, hv.x, fmaf(w2.y, hv.y, fmaf(w2.z, hv.z, fmaf(w2.w, hv.w, g2))));
            g3 = fmaf(w3.x, hv.x, fmaf(w3.y, hv.y, fmaf(w3.z, hv.z, fmaf(w3.w, hv.w, g3))));
        }
        float ig = sigmoidf_dev(g0);
        float fg = sigmoidf_dev(g1);
        float gg = tanhf(g2);
        float og = sigmoidf_dev(g3);
        c = fg * c + ig * gg;
        float hn = og * tanhf(c);

        __syncthreads();           // everyone done reading old h
        h[tid] = hn;
        outb[(size_t)(row0 + t) * 512 + colOff + tid] = hn;
        __syncthreads();           // new h visible before next step's reads
    }
}

// ---------- phase 4: time maxpool + cosine similarity ----------
__global__ __launch_bounds__(512)
void finalize_kernel(const float* __restrict__ q_out, const float* __restrict__ r_out,
                     float* __restrict__ out) {
    __shared__ float red[512];
    int tid = threadIdx.x;

    float qv = -3.4e38f, pv = -3.4e38f, nv = -3.4e38f;
    for (int t = 0; t < QT; ++t) qv = fmaxf(qv, q_out[(size_t)t * 512 + tid]);
    for (int t = 0; t < 32; ++t) pv = fmaxf(pv, r_out[(size_t)t * 512 + tid]);
    for (int t = 32; t < 64; ++t) nv = fmaxf(nv, r_out[(size_t)t * 512 + tid]);

    auto rsum = [&](float v) -> float {
        red[tid] = v;
        __syncthreads();
        for (int sft = 256; sft > 0; sft >>= 1) {
            if (tid < sft) red[tid] += red[tid + sft];
            __syncthreads();
        }
        float r = red[0];
        __syncthreads();
        return r;
    };

    float qq = rsum(qv * qv);
    float pp = rsum(pv * pv);
    float nn = rsum(nv * nv);
    float qp = rsum(qv * pv);
    float qn = rsum(qv * nv);

    if (tid == 0) {
        out[0] = qp / (sqrtf(qq) * sqrtf(pp));
        out[1] = qn / (sqrtf(qq) * sqrtf(nn));
    }
}

// ---------- host ----------
extern "C" void kernel_launch(void* const* d_in, const int* in_sizes, int n_in,
                              void* d_out, int out_size, void* d_ws, size_t ws_size,
                              hipStream_t stream) {
    const int*   question = (const int*)d_in[0];
    const int*   prel     = (const int*)d_in[1];
    const int*   pwrel    = (const int*)d_in[2];
    const int*   nrel     = (const int*)d_in[3];
    const int*   nwrel    = (const int*)d_in[4];
    const float* qemb     = (const float*)d_in[5];
    const float* remb     = (const float*)d_in[6];

    // q_lstm / r_lstm arrive as 8 flattened pytree leaves each, in jax's
    // sorted-key order: bwd{W_hh,W_ih,b_hh,b_ih}, fwd{W_hh,W_ih,b_hh,b_ih}.
    // Resolve by element count for robustness to interleaving.
    const float* Whh_q[2] = {0, 0}; const float* Wih_q[2] = {0, 0}; const float* bias_q[4] = {0, 0, 0, 0};
    const float* Whh_r[2] = {0, 0}; const float* Wih_r[2] = {0, 0}; const float* bias_r[4] = {0, 0, 0, 0};
    auto scan8 = [&](int base, const float** Whh, const float** Wih, const float** bias) {
        int a = 0, b = 0, c = 0;
        for (int i = base; i < base + 8 && i < n_in; ++i) {
            int sz = in_sizes[i];
            if      (sz == GATE * HID) { if (a < 2) Whh[a++] = (const float*)d_in[i]; }
            else if (sz == GATE * EMB) { if (b < 2) Wih[b++] = (const float*)d_in[i]; }
            else if (sz == GATE)       { if (c < 4) bias[c++] = (const float*)d_in[i]; }
        }
    };
    scan8(7,  Whh_q, Wih_q, bias_q);
    scan8(15, Whh_r, Wih_r, bias_r);
    // index 0 = bwd, 1 = fwd; biases: [0]=bwd b_hh, [1]=bwd b_ih, [2]=fwd b_hh, [3]=fwd b_ih
    const float* whh_qf = Whh_q[1], *whh_qb = Whh_q[0];
    const float* whh_rf = Whh_r[1], *whh_rb = Whh_r[0];
    const float* wih_qf = Wih_q[1], *wih_qb = Wih_q[0];
    const float* wih_rf = Wih_r[1], *wih_rb = Wih_r[0];

    // workspace layout (~4 MB)
    unsigned short* xq   = (unsigned short*)d_ws;              // 64*320 bf16
    unsigned short* xr   = xq + (size_t)QT * KPAD;             // 64*320 bf16
    unsigned short* wt   = xr + (size_t)RROWS * KPAD;          // 4*64*10*32*16 bf16 (fragments)
    float*          bsum = (float*)(wt + (size_t)4 * KPAD * GATE);   // 4*1024 f32
    float*          xg   = bsum + (size_t)4 * GATE;            // 4*64*1024 f32
    float*          qout = xg + (size_t)4 * 64 * GATE;         // 64*512 f32
    float*          rout = qout + (size_t)QT * 512;            // 64*512 f32

    gather_kernel<<<QT + RROWS, KPAD, 0, stream>>>(question, prel, pwrel, nrel, nwrel,
                                                   qemb, remb, xq, xr);
    wswz_kernel<<<dim3(64, 4), KT * 32, 0, stream>>>(wih_qf, wih_qb, wih_rf, wih_rb, wt);
    bias_kernel<<<4, 256, 0, stream>>>(bias_q[3], bias_q[2],   // g0: q fwd  b_ih + b_hh
                                       bias_q[1], bias_q[0],   // g1: q bwd
                                       bias_r[3], bias_r[2],   // g2: r fwd
                                       bias_r[1], bias_r[0],   // g3: r bwd
                                       bsum);
    xg_gemm_kernel<<<dim3(64, 4, 4), 32, 0, stream>>>((const __bf16*)xq, (const __bf16*)xr,
                                                      (const __bf16*)wt, bsum, xg);
    lstm_scan_kernel<<<10, 256, 0, stream>>>(xg, whh_qf, whh_qb, whh_rf, whh_rb, qout, rout);
    finalize_kernel<<<1, 512, 0, stream>>>(qout, rout, (float*)d_out);
}
